// InterestTransformerDecoder_64982855188796
// MI455X (gfx1250) — matrix operand order repaired and checked by reading, not compiled, not run on hardware
//
#include <hip/hip_runtime.h>
#include <hip/hip_bf16.h>

// ---------------------------------------------------------------------------
// Types for CDNA5 WMMA (wave32): f16 A/B fragments (16 halves / lane),
// fp32 C/D fragments (8 floats / lane).
// ---------------------------------------------------------------------------
typedef __attribute__((ext_vector_type(16))) _Float16 v16h;
typedef __attribute__((ext_vector_type(8)))  _Float16 v8h;
typedef __attribute__((ext_vector_type(4)))  _Float16 v4h;
typedef __attribute__((ext_vector_type(8)))  float    v8f;

#define BB 512
#define LL 256
#define DD 128

// LDS strides (in _Float16 elements); row strides kept 16B-aligned.
#define SH_STRIDE  136   // 256 x 136 f16  = 69632 B
#define SHT_STRIDE 264   // 128 x 264 f16  = 67584 B

// dynamic-LDS byte offsets for kernel 1
#define OFF_SH    0
#define OFF_SHT   69632
#define OFF_STG   137216          // 8 waves * 16*32 f16 = 8192 B
#define OFF_INV   145408          // float[256]
#define OFF_M     146432          // float[256]
#define OFF_LM    147456          // float[256]
#define OFF_HLAST 148480          // float[128]
#define K1_LDS    148992

__device__ __forceinline__ v8f v8f_zero() {
    v8f v;
#pragma unroll
    for (int i = 0; i < 8; ++i) v[i] = 0.0f;
    return v;
}

__device__ __forceinline__ v16h cat8(v8h lo, v8h hi) {
    return __builtin_shufflevector(lo, hi, 0,1,2,3,4,5,6,7,8,9,10,11,12,13,14,15);
}

// A fragment, 16x32 f16 (MxK), row-major source. Lane layout per ISA:
// lanes 0-15 hold row M=lane, K = k0+{0..7} and k0+16+{0..7};
// lanes 16-31 hold row M=lane-16, K = k0+8+{0..7} and k0+24+{0..7}.
__device__ __forceinline__ v16h load_frag_a(const _Float16* base, int stride,
                                            int lane, int k0) {
    const int M = lane & 15, half = lane >> 4;
    const _Float16* p = base + M * stride + k0 + 8 * half;
    v8h lo = *(const v8h*)p;
    v8h hi = *(const v8h*)(p + 16);
    return cat8(lo, hi);
}

// B fragment, 32x16 f16 (KxN), where memory holds B transposed row-major
// (i.e. base[(row0+n)*stride + col0 + k] == B[k][n]). Lane n=lane&15 reads
// K = col0+8*(lane>>4)+{0..7} and col0+16+8*(lane>>4)+{0..7}.
__device__ __forceinline__ v16h load_frag_b(const _Float16* base, int stride,
                                            int lane, int row0, int col0) {
    const int nn = lane & 15, half = lane >> 4;
    const _Float16* p = base + (row0 + nn) * stride + col0 + 8 * half;
    v8h lo = *(const v8h*)p;
    v8h hi = *(const v8h*)(p + 16);
    return cat8(lo, hi);
}

__device__ __forceinline__ v8f wmma_f16(v16h a, v16h b, v8f c) {
    // D = A x B + C ; fp32 accumulate
    return __builtin_amdgcn_wmma_f32_16x16x32_f16(false, a, false, b,
                                                  (short)0, c, false, false);
}

// ===========================================================================
// Kernel 1: per-batch self cosine-similarity attention -> h_last[b][128]
// ===========================================================================
__global__ __launch_bounds__(256) void k1_self_attn(
    const int* __restrict__ seq, const int* __restrict__ lastseq,
    const float* __restrict__ hist1, float* __restrict__ hlast) {
    extern __shared__ char smem[];
    _Float16* sH     = (_Float16*)(smem + OFF_SH);
    _Float16* sHT    = (_Float16*)(smem + OFF_SHT);
    _Float16* sStg   = (_Float16*)(smem + OFF_STG);
    float*    sInv   = (float*)(smem + OFF_INV);
    float*    sM     = (float*)(smem + OFF_M);
    float*    sLM    = (float*)(smem + OFF_LM);
    float*    sHlast = (float*)(smem + OFF_HLAST);

    const int b    = blockIdx.x;
    const int tid  = threadIdx.x;
    const int wave = tid >> 5;
    const int lane = tid & 31;

    // ---- masks ----
    {
        const int s  = seq[b * LL + tid];
        const int ls = lastseq[b * LL + tid];
        const float m = (s != 0) ? 1.0f : 0.0f;
        sM[tid]  = m;
        sLM[tid] = m * (float)ls;
    }
    if (tid < DD) sHlast[tid] = 0.0f;

    // ---- load H1, convert to f16 (row-major + transposed), row norms ----
    const float* hb = hist1 + (size_t)b * LL * DD;
#pragma unroll 4
    for (int rr = 0; rr < 32; ++rr) {
        const int r = wave * 32 + rr;
        const float4 v = *(const float4*)(hb + r * DD + lane * 4);
        float sq = v.x * v.x + v.y * v.y + v.z * v.z + v.w * v.w;
        const _Float16 e0 = (_Float16)v.x, e1 = (_Float16)v.y;
        const _Float16 e2 = (_Float16)v.z, e3 = (_Float16)v.w;
        v4h pk = {e0, e1, e2, e3};
        *(v4h*)(sH + r * SH_STRIDE + lane * 4) = pk;
        const int c = lane * 4;
        sHT[(c + 0) * SHT_STRIDE + r] = e0;
        sHT[(c + 1) * SHT_STRIDE + r] = e1;
        sHT[(c + 2) * SHT_STRIDE + r] = e2;
        sHT[(c + 3) * SHT_STRIDE + r] = e3;
#pragma unroll
        for (int off = 16; off > 0; off >>= 1) sq += __shfl_xor(sq, off, 32);
        if (lane == 0) sInv[r] = rsqrtf(sq + 0.01f);
    }
    __syncthreads();

    _Float16* stg = sStg + wave * (16 * 32);
    const int nn = lane & 15, half = lane >> 4;

    // Each wave handles two 16-row i-tiles.
    for (int itp = 0; itp < 2; ++itp) {
        const int i0 = (wave + itp * 8) * 16;

        v16h a4[4];
#pragma unroll
        for (int k = 0; k < 4; ++k)
            a4[k] = load_frag_a(sH + i0 * SH_STRIDE, SH_STRIDE, lane, 32 * k);

        float invi[8];
#pragma unroll
        for (int g = 0; g < 8; ++g) invi[g] = sInv[i0 + g + 8 * half];

        v8f hacc[8];
#pragma unroll
        for (int t = 0; t < 8; ++t) hacc[t] = v8f_zero();
        v8f wacc = v8f_zero();

        // --- software pipeline: preload B fragments for jt=0 ---
        v16h bf[4], bfn[4];
#pragma unroll
        for (int k = 0; k < 4; ++k)
            bf[k] = load_frag_b(sH, SH_STRIDE, lane, 0, 32 * k);

        for (int jt = 0; jt < 16; ++jt) {
            const int j0 = jt * 16;

            // prefetch next j-tile's B fragments; overlaps with WMMA chain
            if (jt < 15) {
#pragma unroll
                for (int k = 0; k < 4; ++k)
                    bfn[k] = load_frag_b(sH, SH_STRIDE, lane, j0 + 16, 32 * k);
            }

            // ---- S tile: 4 x WMMA over K = 128 ----
            v8f c = v8f_zero();
#pragma unroll
            for (int k = 0; k < 4; ++k) c = wmma_f16(a4[k], bf[k], c);

            // ---- cos -> exp -> mask ; stage w as f16 ; row-sum partials ----
            const float invj = sInv[j0 + nn];
            const float mj   = sM[j0 + nn];
            const int cpar   = (jt & 1) * 16 + nn;
#pragma unroll
            for (int g = 0; g < 8; ++g) {
                const float wv = __expf(c[g] * invi[g] * invj) * mj;
                wacc[g] += wv;
                stg[(g + 8 * half) * 32 + cpar] = (_Float16)wv;
            }

            // ---- every j-pair: h_self += w_tile(16x32) @ H1(32x128) ----
            if (jt & 1) {
                asm volatile("s_wait_dscnt 0" ::: "memory");
                const int j0p = (jt - 1) * 16;
                v16h a2 = load_frag_a(stg, 32, lane, 0);
                // pairs of independent WMMAs; two loads in flight per step
#pragma unroll
                for (int t = 0; t < 8; t += 2) {
                    v16h b2a = load_frag_b(sHT, SHT_STRIDE, lane, t * 16, j0p);
                    v16h b2b = load_frag_b(sHT, SHT_STRIDE, lane, (t + 1) * 16, j0p);
                    hacc[t]     = wmma_f16(a2, b2a, hacc[t]);
                    hacc[t + 1] = wmma_f16(a2, b2b, hacc[t + 1]);
                }
            }

#pragma unroll
            for (int k = 0; k < 4; ++k) bf[k] = bfn[k];
        }

        // ---- reduce wsum across each 16-lane group ----
#pragma unroll
        for (int off = 1; off < 16; off <<= 1) {
#pragma unroll
            for (int g = 0; g < 8; ++g) wacc[g] += __shfl_xor(wacc[g], off, 32);
        }

        // ---- residual + query mask + 'last' pooling into sHlast ----
#pragma unroll
        for (int g = 0; g < 8; ++g) {
            const int i = i0 + g + 8 * half;
            const float lm = sLM[i];
            if (lm != 0.0f) {
                const float invw = 1.0f / wacc[g];
#pragma unroll
                for (int t = 0; t < 8; ++t) {
                    const int d = t * 16 + nn;
                    const float hv = (float)sH[i * SH_STRIDE + d];
                    atomicAdd(&sHlast[d], lm * (hacc[t][g] * invw + hv));
                }
            }
        }
    }
    __syncthreads();
    if (tid < DD) hlast[(size_t)b * DD + tid] = sHlast[tid];
}

// ===========================================================================
// Kernel 2: cross cosine attention: h_last attends over hist_2 -> h[b][128]
// ===========================================================================
__global__ __launch_bounds__(256) void k2_cross_attn(
    const int* __restrict__ seq1, const float* __restrict__ hist2,
    const float* __restrict__ hlast, float* __restrict__ hout) {
    __shared__ float sHL[DD];
    __shared__ float sW2[LL];
    __shared__ float sRed[2];

    const int b = blockIdx.x, tid = threadIdx.x;
    const int wave = tid >> 5, lane = tid & 31;

    if (tid < DD) sHL[tid] = hlast[(size_t)b * DD + tid];
    __syncthreads();

    if (wave == 0) {
        const float4 q = *(const float4*)(sHL + lane * 4);
        float sq = q.x * q.x + q.y * q.y + q.z * q.z + q.w * q.w;
#pragma unroll
        for (int off = 16; off > 0; off >>= 1) sq += __shfl_xor(sq, off, 32);
        if (lane == 0) sRed[0] = rsqrtf(sq + 0.01f);
    }
    __syncthreads();
    const float invq = sRed[0];
    const float* h2b = hist2 + (size_t)b * LL * DD;

#pragma unroll 4
    for (int rr = 0; rr < 32; ++rr) {
        const int k = wave * 32 + rr;
        const float4 v = *(const float4*)(h2b + k * DD + lane * 4);
        const float4 q = *(const float4*)(sHL + lane * 4);
        float dp = v.x * q.x + v.y * q.y + v.z * q.z + v.w * q.w;
        float sq = v.x * v.x + v.y * v.y + v.z * v.z + v.w * v.w;
#pragma unroll
        for (int off = 16; off > 0; off >>= 1) {
            dp += __shfl_xor(dp, off, 32);
            sq += __shfl_xor(sq, off, 32);
        }
        if (lane == 0) {
            const float m1 = (seq1[b * LL + k] != 0) ? 1.0f : 0.0f;
            sW2[k] = __expf(dp * rsqrtf(sq + 0.01f) * invq) * m1;
        }
    }
    __syncthreads();

    if (wave == 0) {
        float s = 0.0f;
        for (int k = lane; k < LL; k += 32) s += sW2[k];
#pragma unroll
        for (int off = 16; off > 0; off >>= 1) s += __shfl_xor(s, off, 32);
        if (lane == 0) sRed[1] = s;
    }
    __syncthreads();
    const float inv_wsum = 1.0f / sRed[1];

    if (tid < DD) {
        float acc = 0.0f;
        for (int k = 0; k < LL; ++k) acc += sW2[k] * h2b[k * DD + tid];
        hout[(size_t)b * DD + tid] = acc * inv_wsum;
    }
}

// ===========================================================================
// Kernel 3: out = relu(H @ W), H = [512,128], W = [128,128] -> WMMA GEMM
// Each block handles 128 batch rows; 8 waves -> 8 M-tiles of 16.
// ===========================================================================
#define K3_STRIDE 136
#define K3_LDS (2 * 128 * K3_STRIDE * 2)   // sA + sWT, f16

__global__ __launch_bounds__(256) void k3_out_gemm(
    const float* __restrict__ h, const float* __restrict__ W,
    float* __restrict__ out) {
    extern __shared__ char smem[];
    _Float16* sA  = (_Float16*)smem;                              // 128 x 136
    _Float16* sWT = (_Float16*)(smem + 128 * K3_STRIDE * 2);      // 128 x 136 (W^T)

    const int tid = threadIdx.x, wave = tid >> 5, lane = tid & 31;
    const int b0 = blockIdx.x * 128;

    for (int e = tid; e < 128 * 128; e += 256) {
        const int r = e >> 7, c = e & 127;
        sA[r * K3_STRIDE + c]  = (_Float16)h[(size_t)(b0 + r) * DD + c];
        sWT[c * K3_STRIDE + r] = (_Float16)W[e];   // W[r][c] -> sWT[c][r]
    }
    __syncthreads();

    const int m0 = wave * 16;
    const int nn = lane & 15, half = lane >> 4;

    v16h a4[4];
#pragma unroll
    for (int k = 0; k < 4; ++k)
        a4[k] = load_frag_a(sA + m0 * K3_STRIDE, K3_STRIDE, lane, 32 * k);

#pragma unroll
    for (int t = 0; t < 8; ++t) {
        // batch all 4 B-fragment loads ahead of the WMMA chain
        v16h bf4[4];
#pragma unroll
        for (int k = 0; k < 4; ++k)
            bf4[k] = load_frag_b(sWT, K3_STRIDE, lane, t * 16, 32 * k);
        v8f c = v8f_zero();
#pragma unroll
        for (int k = 0; k < 4; ++k) c = wmma_f16(a4[k], bf4[k], c);
#pragma unroll
        for (int g = 0; g < 8; ++g) {
            const int r = b0 + m0 + g + 8 * half;
            const int col = t * 16 + nn;
            out[(size_t)r * DD + col] = fmaxf(c[g], 0.0f);
        }
    }
}

// ===========================================================================
extern "C" void kernel_launch(void* const* d_in, const int* in_sizes, int n_in,
                              void* d_out, int out_size, void* d_ws, size_t ws_size,
                              hipStream_t stream) {
    const int*   seq     = (const int*)d_in[0];    // [512,256]
    const int*   lastseq = (const int*)d_in[1];    // [512,256]
    const int*   seq1    = (const int*)d_in[2];    // [512,256]
    const float* hist1   = (const float*)d_in[3];  // [512,256,128]
    const float* hist2   = (const float*)d_in[4];  // [512,256,128]
    const float* W       = (const float*)d_in[5];  // [128,128]
    float* out = (float*)d_out;                    // [512,1,128]

    float* ws_hlast = (float*)d_ws;                          // 512*128 f32
    float* ws_h     = (float*)((char*)d_ws + BB * DD * 4);   // 512*128 f32

    (void)in_sizes; (void)n_in; (void)out_size; (void)ws_size;

    // Allow >64KB dynamic LDS (WGP has 320KB).
    hipFuncSetAttribute((const void*)k1_self_attn,
                        hipFuncAttributeMaxDynamicSharedMemorySize, K1_LDS);
    hipFuncSetAttribute((const void*)k3_out_gemm,
                        hipFuncAttributeMaxDynamicSharedMemorySize, K3_LDS);

    k1_self_attn<<<BB, 256, K1_LDS, stream>>>(seq, lastseq, hist1, ws_hlast);
    k2_cross_attn<<<BB, 256, 0, stream>>>(seq1, hist2, ws_hlast, ws_h);
    k3_out_gemm<<<BB / 128, 256, K3_LDS, stream>>>(ws_h, W, out);
}
